// GraphAttentionLayer_rel_25022479466557
// MI455X (gfx1250) — matrix operand order repaired and checked by reading, not compiled, not run on hardware
//
#include <hip/hip_runtime.h>
#include <hip/hip_bf16.h>

typedef __attribute__((ext_vector_type(16))) _Float16 v16h;
typedef __attribute__((ext_vector_type(8)))  float    v8f;

#define NN     10000
#define DD     128
#define NRELC  237
#define RR     5000
#define EE     160000
#define NGRP   313            // ceil(NN/32) j-groups, B padded to 313*32 rows
#define LRELU_ALPHA 0.2f

__device__ __forceinline__ unsigned pack2h(float a, float b) {
  union { _Float16 h[2]; unsigned u; } p;
  p.h[0] = (_Float16)a; p.h[1] = (_Float16)b;
  return p.u;
}

// ---------------------------------------------------------------------------
// K1: seq_fts = input @ W^T, scattered directly into WMMA B-operand order:
// element (row i, col d):  g=i/32, k=i%32, lane=(d&15)+16*(k>=16), t=d>>4,
// q=(k&15)>>1, halfsel=k&1.  Flat f16 index = ((((g*8+t)*32+lane)*8+q)*2+halfsel.
// Each hot-loop lane then reads its 8 B VGPRs as 2 contiguous b128 loads.
// ---------------------------------------------------------------------------
__global__ void k_seqfts_pack(const float* __restrict__ input,
                              const float* __restrict__ W,
                              _Float16* __restrict__ packedB) {
  __shared__ float srow[DD];
  const int i = blockIdx.x;
  const int d = threadIdx.x;
  srow[d] = input[(long)i * DD + d];
  __syncthreads();
  const float* wr = W + (long)d * DD;
  float s = 0.f;
#pragma unroll 8
  for (int k = 0; k < DD; ++k) s = fmaf(srow[k], wr[k], s);
  const int g = i >> 5, k = i & 31;
  const int lane = (d & 15) + ((k & 16) ? 16 : 0);
  const int t = d >> 4;
  const int q = (k & 15) >> 1;
  const int hs = k & 1;
  packedB[((((long)g * 8 + t) * 32 + lane) * 8 + q) * 2 + hs] = (_Float16)s;
}

// ---------------------------------------------------------------------------
// K2a: rel_logits[r] = dot(rel[r,:], W_rel)
// ---------------------------------------------------------------------------
__global__ void k_rel_logits(const float* __restrict__ rel,
                             const float* __restrict__ W_rel,
                             float* __restrict__ rl) {
  const int r = blockIdx.x * blockDim.x + threadIdx.x;
  if (r >= RR) return;
  const float* rr = rel + (long)r * NRELC;
  float s = 0.f;
  for (int k = 0; k < NRELC; ++k) s = fmaf(rr[k], W_rel[k], s);
  rl[r] = s;
}

// ---------------------------------------------------------------------------
// K2b: zero-fill (float4 grid-stride) — used for logits and packedB pad.
// ---------------------------------------------------------------------------
__global__ void k_zero4(float4* __restrict__ p, long n4) {
  long i = (long)blockIdx.x * blockDim.x + threadIdx.x;
  const long stride = (long)gridDim.x * blockDim.x;
  const float4 z = make_float4(0.f, 0.f, 0.f, 0.f);
  for (; i < n4; i += stride) p[i] = z;
}

// ---------------------------------------------------------------------------
// K2c: edge_vals = max over K=2 of rel_logits[edge_rel]; symmetric scatter.
// ---------------------------------------------------------------------------
__global__ void k_scatter(const int* __restrict__ edge_index,
                          const int* __restrict__ edge_rel,
                          const float* __restrict__ rl,
                          float* __restrict__ logits) {
  const int e = blockIdx.x * blockDim.x + threadIdx.x;
  if (e >= EE) return;
  const int e1 = edge_index[(long)e * 2 + 0];
  const int e2 = edge_index[(long)e * 2 + 1];
  const float v = fmaxf(rl[edge_rel[(long)e * 2 + 0]],
                        rl[edge_rel[(long)e * 2 + 1]]);
  logits[(long)e1 * NN + e2] = v;
  logits[(long)e2 * NN + e1] = v;
}

// ---------------------------------------------------------------------------
// K3: fused pass.  t_ij = exp(leaky_relu(logits)+adj) streamed once;
// numerator via v_wmma_f32_16x16x32_f16, denominator via row sums.
// 625 blocks x 16 rows; 8 waves split the 313 j-chunks (K=32 each).
// ---------------------------------------------------------------------------
__global__ __launch_bounds__(256)
void k_gat_main(const float* __restrict__ logits,
                const float* __restrict__ adj,
                const uint4* __restrict__ packedB4, // B operand order
                const float* __restrict__ bias,
                float* __restrict__ out) {
  __shared__ __align__(16) unsigned char smemRaw[64 * 1024]; // tTiles then C
  __shared__ float smemRS[8][16];
  __shared__ float smemRSt[16];

  const int tid  = threadIdx.x;
  const int w    = tid >> 5;
  const int lane = tid & 31;
  const int rowbase = blockIdx.x * 16;

  _Float16* tTile = (_Float16*)smemRaw + (size_t)w * 512;   // 16x32 halves
  uint2*       tTileW = (uint2*)tTile;                      // b64 store view
  const uint4* tTileA = (const uint4*)tTile;                // b128 A-load view

  if (lane < 16) smemRS[w][lane] = 0.f;

  v8f c[8] = {};
  float partial4[4] = {0.f, 0.f, 0.f, 0.f};

  const int myrow = lane & 15;
  const int hi    = lane >> 4;            // 0|1
  const int rquad = (lane >> 3) & 3;      // staging: row offset within quad? no:
  // staging mapping: lane covers (row q*4 + (lane>>3)&3? ) — use lane>>3 in 0..3
  const int rsub  = lane >> 3;            // 0..3 : row within 4-row group
  const int csub  = (lane & 7) * 4;       // 0,4,..,28 : col within chunk
  (void)rquad;

  // A-operand b128 indices (contiguous k = 16*hi + s, same order as packed B)
  const int aIdx0 = myrow * 4 + hi * 2;   // K 0..7  (hi: 16..23)
  const int aIdx1 = aIdx0 + 1;            // K 8..15 (hi: 24..31)

  // staging LDS b64 slot per (q): (q*4+rsub)*8 + (lane&7)
  const int sBase = rsub * 8 + (lane & 7);

  for (int ch = w; ch < NGRP - 1; ch += 8) {   // guard-free main chunks
    const int jbase = ch * 32;

    // ---- stage t tile: 4x (b128 logits + b128 adj) per lane ----
#pragma unroll
    for (int q = 0; q < 4; ++q) {
      const long row = rowbase + q * 4 + rsub;
      const long base = (row * (long)NN + jbase + csub) >> 2;  // float4 index
      const float4 lg = ((const float4*)logits)[base];
      const float4 av = ((const float4*)adj)[base];
      const float t0 = __expf(((lg.x > 0.f) ? lg.x : LRELU_ALPHA * lg.x) + av.x);
      const float t1 = __expf(((lg.y > 0.f) ? lg.y : LRELU_ALPHA * lg.y) + av.y);
      const float t2 = __expf(((lg.z > 0.f) ? lg.z : LRELU_ALPHA * lg.z) + av.z);
      const float t3 = __expf(((lg.w > 0.f) ? lg.w : LRELU_ALPHA * lg.w) + av.w);
      partial4[q] += (t0 + t1) + (t2 + t3);
      tTileW[q * 32 + sBase] = make_uint2(pack2h(t0, t1), pack2h(t2, t3));
    }

    // prefetch next chunk for this wave (gfx1250 global_prefetch_b8)
    const int jnext = jbase + 8 * 32;
    if (jnext < NN) {
      const long pbase = (long)(rowbase + myrow) * NN + jnext + hi * 16;
      __builtin_prefetch(logits + pbase, 0, 0);
      __builtin_prefetch(adj + pbase, 0, 0);
    }

    asm volatile("s_wait_dscnt 0" ::: "memory");  // wave-local LDS RAW fence

    // ---- A operand: two b128 LDS loads ----
    union { v16h v; uint4 u4[2]; } A;
    A.u4[0] = tTileA[aIdx0];
    A.u4[1] = tTileA[aIdx1];

    // ---- B operands: 2 contiguous b128 global loads per tile; 8 WMMAs ----
    const long bBase = ((long)ch * 256 + lane) * 2;
#pragma unroll
    for (int t = 0; t < 8; ++t) {
      union { v16h v; uint4 u4[2]; } B;
      B.u4[0] = packedB4[bBase + t * 64];
      B.u4[1] = packedB4[bBase + t * 64 + 1];
      c[t] = __builtin_amdgcn_wmma_f32_16x16x32_f16(
          false, A.v, false, B.v, (short)0, c[t], false, false);
    }
  }

  // ---- ragged tail chunk (jbase=9984, 16 valid cols), wave 0 only ----
  if (w == 0) {
    const int ch = NGRP - 1;
    const int jbase = ch * 32;
    const bool valid = (lane & 7) < 4;    // csub 0..12 -> cols 9984..9999
#pragma unroll
    for (int q = 0; q < 4; ++q) {
      const long row = rowbase + q * 4 + rsub;
      float t0 = 0.f, t1 = 0.f, t2 = 0.f, t3 = 0.f;
      if (valid) {
        const long base = (row * (long)NN + jbase + csub) >> 2;
        const float4 lg = ((const float4*)logits)[base];
        const float4 av = ((const float4*)adj)[base];
        t0 = __expf(((lg.x > 0.f) ? lg.x : LRELU_ALPHA * lg.x) + av.x);
        t1 = __expf(((lg.y > 0.f) ? lg.y : LRELU_ALPHA * lg.y) + av.y);
        t2 = __expf(((lg.z > 0.f) ? lg.z : LRELU_ALPHA * lg.z) + av.z);
        t3 = __expf(((lg.w > 0.f) ? lg.w : LRELU_ALPHA * lg.w) + av.w);
      }
      partial4[q] += (t0 + t1) + (t2 + t3);
      tTileW[q * 32 + sBase] = make_uint2(pack2h(t0, t1), pack2h(t2, t3));
    }
    asm volatile("s_wait_dscnt 0" ::: "memory");
    union { v16h v; uint4 u4[2]; } A;
    A.u4[0] = tTileA[aIdx0];
    A.u4[1] = tTileA[aIdx1];
    const long bBase = ((long)ch * 256 + lane) * 2;  // pad rows are zeroed
#pragma unroll
    for (int t = 0; t < 8; ++t) {
      union { v16h v; uint4 u4[2]; } B;
      B.u4[0] = packedB4[bBase + t * 64];
      B.u4[1] = packedB4[bBase + t * 64 + 1];
      c[t] = __builtin_amdgcn_wmma_f32_16x16x32_f16(
          false, A.v, false, B.v, (short)0, c[t], false, false);
    }
  }

  // ---- row-sum partials -> per-wave LDS slots (static reg indices) ----
  atomicAdd(&smemRS[w][0 * 4 + rsub], partial4[0]);
  atomicAdd(&smemRS[w][1 * 4 + rsub], partial4[1]);
  atomicAdd(&smemRS[w][2 * 4 + rsub], partial4[2]);
  atomicAdd(&smemRS[w][3 * 4 + rsub], partial4[3]);

  __syncthreads();  // tTile region now dead -> reuse smemRaw for C tiles

  // C/D layout: VGPR r of tile t, lane l -> row r + 8*hi, col t*16 + (l&15)
  float* cBuf = (float*)smemRaw + (size_t)w * 2048;
#pragma unroll
  for (int t = 0; t < 8; ++t)
#pragma unroll
    for (int r = 0; r < 8; ++r)
      cBuf[(r + hi * 8) * DD + t * 16 + myrow] = c[t][r];

  if (tid < 16) {
    float s = 0.f;
#pragma unroll
    for (int wv = 0; wv < 8; ++wv) s += smemRS[wv][tid];
    smemRSt[tid] = s;
  }
  __syncthreads();

  const float* cAll = (const float*)smemRaw;
  for (int i = tid; i < 16 * DD; i += 256) {
    float v = 0.f;
#pragma unroll
    for (int wv = 0; wv < 8; ++wv) v += cAll[wv * 2048 + i];
    const int r = i >> 7, col = i & 127;
    const float x = v / smemRSt[r] + bias[col];
    out[(long)(rowbase + r) * DD + col] = (x > 0.f) ? x : (__expf(x) - 1.f);
  }
}

// ---------------------------------------------------------------------------
extern "C" void kernel_launch(void* const* d_in, const int* in_sizes, int n_in,
                              void* d_out, int out_size, void* d_ws, size_t ws_size,
                              hipStream_t stream) {
  (void)in_sizes; (void)n_in; (void)out_size; (void)ws_size;
  const float* input      = (const float*)d_in[0];
  const float* rel        = (const float*)d_in[1];
  const float* adj        = (const float*)d_in[2];
  const float* W          = (const float*)d_in[3];
  const float* W_rel      = (const float*)d_in[4];
  const float* bias       = (const float*)d_in[5];
  const int*   edge_index = (const int*)d_in[6];
  const int*   edge_rel   = (const int*)d_in[7];
  float* out = (float*)d_out;

  char* ws = (char*)d_ws;
  float* logits = (float*)ws;                              // N*N f32 = 400 MB
  size_t off = (size_t)NN * NN * sizeof(float);
  _Float16* packedB = (_Float16*)(ws + off);               // NGRP*8*32*8 uints
  const size_t packedBytes = (size_t)NGRP * 8 * 32 * 8 * 4;
  off += packedBytes;
  float* rl = (float*)(ws + off);                          // R f32

  // zero B-operand buffer (pad rows must be 0.0h), then fill it
  k_zero4<<<512, 256, 0, stream>>>((float4*)packedB, (long)(packedBytes / 16));
  k_seqfts_pack<<<NN, DD, 0, stream>>>(input, W, packedB);
  k_rel_logits<<<(RR + 255) / 256, 256, 0, stream>>>(rel, W_rel, rl);
  k_zero4<<<4096, 256, 0, stream>>>((float4*)logits, (long)NN * NN / 4);
  k_scatter<<<(EE + 255) / 256, 256, 0, stream>>>(edge_index, edge_rel, rl, logits);
  k_gat_main<<<NN / 16, 256, 0, stream>>>(logits, adj,
                                          (const uint4*)packedB, bias, out);
}
